// RNN_65489661329919
// MI455X (gfx1250) — matrix operand order repaired and checked by reading, not compiled, not run on hardware
//
#include <hip/hip_runtime.h>
#include <cmath>

// ---------------- problem dims (fixed by reference) ----------------
#define B_ 64
#define T_ 512
#define I_ 1024
#define H_ 1024

typedef __attribute__((ext_vector_type(16))) __bf16 v16bf;
typedef __attribute__((ext_vector_type(8)))  __bf16 v8bf;
typedef __attribute__((ext_vector_type(8)))  float  v8f;
typedef unsigned int u32x4 __attribute__((ext_vector_type(4)));
typedef int          i32x8 __attribute__((ext_vector_type(8)));
typedef int          i32x4 __attribute__((ext_vector_type(4)));

// ---------------- CDNA5 feature probes (compile-safe) ----------------
#if defined(__gfx1250__) && __has_builtin(__builtin_amdgcn_global_load_async_to_lds_b128)
#define HAS_ASYNC_LDS 1
#else
#define HAS_ASYNC_LDS 0
#endif

#if defined(__gfx1250__) && __has_builtin(__builtin_amdgcn_tensor_load_to_lds) && \
    __has_builtin(__builtin_amdgcn_s_wait_tensorcnt)
#define HAS_TDM 1
#else
#define HAS_TDM 0
#endif

// async 16-byte global->LDS copy (falls back to register copy)
__device__ __forceinline__ void async_copy16(const __bf16* g, __bf16* l) {
#if HAS_ASYNC_LDS
  __builtin_assume(g != nullptr);
  __builtin_assume(l != nullptr);
  __builtin_amdgcn_global_load_async_to_lds_b128(
      (__attribute__((address_space(1))) i32x4*)(void*)(g),
      (__attribute__((address_space(3))) i32x4*)(void*)(l), 0, 0);
#else
  *(v8bf*)l = *(const v8bf*)g;
#endif
}

__device__ __forceinline__ void async_wait0() {
#if HAS_ASYNC_LDS
#if __has_builtin(__builtin_amdgcn_s_wait_asynccnt)
  __builtin_amdgcn_s_wait_asynccnt(0);
#else
  asm volatile("s_wait_asynccnt 0x0" ::: "memory");
#endif
#endif
}

// =====================================================================
// Prep kernels
// =====================================================================
__global__ void k_transpose_bf16(const float* __restrict__ w,
                                 __bf16* __restrict__ wt,
                                 int K, int N) {
  int idx = blockIdx.x * blockDim.x + threadIdx.x;
  if (idx >= K * N) return;
  int n = idx / K;
  int k = idx - n * K;
  wt[idx] = (__bf16)w[(size_t)k * N + n];
}

__global__ void k_hx_to_bf16(const float* __restrict__ hx,
                             __bf16* __restrict__ h0) {
  int idx = blockIdx.x * blockDim.x + threadIdx.x;
  if (idx < B_ * H_) h0[idx] = (__bf16)hx[idx];
}

// A-fragment builder: fp32 row -> bf16 16x32 fragment slice for this lane
__device__ __forceinline__ v16bf load_a_frag(const float* __restrict__ row,
                                             int k0, int kb) {
  const float4 f0 = *(const float4*)(row + k0 + kb);
  const float4 f1 = *(const float4*)(row + k0 + kb + 4);
  const float4 f2 = *(const float4*)(row + k0 + 16 + kb);
  const float4 f3 = *(const float4*)(row + k0 + 16 + kb + 4);
  v16bf a;
  a[0]=(__bf16)f0.x;  a[1]=(__bf16)f0.y;  a[2]=(__bf16)f0.z;  a[3]=(__bf16)f0.w;
  a[4]=(__bf16)f1.x;  a[5]=(__bf16)f1.y;  a[6]=(__bf16)f1.z;  a[7]=(__bf16)f1.w;
  a[8]=(__bf16)f2.x;  a[9]=(__bf16)f2.y;  a[10]=(__bf16)f2.z; a[11]=(__bf16)f2.w;
  a[12]=(__bf16)f3.x; a[13]=(__bf16)f3.y; a[14]=(__bf16)f3.z; a[15]=(__bf16)f3.w;
  return a;
}

// =====================================================================
// Kernel A: x_proj = input @ W_ih + bias_ih + bias_hh  -> d_out (in place)
//   M=B*T=32768, N=H=1024, K=I=1024.
//   Block tile 128(M) x 128(N); 8 waves, each 32x64 (2 Mfrag x 4 Nfrag).
//   B tile (32x128 bf16 = 8KB) double-buffered in LDS via async-to-LDS;
//   k-loop unrolled x2 so buffer selection is static.
// =====================================================================
__global__ __launch_bounds__(256)
void k_gemm_xproj(const float*  __restrict__ x,    // [M][K]
                  const __bf16* __restrict__ wt,   // [N][K]
                  const float*  __restrict__ bih,
                  const float*  __restrict__ bhh,
                  float*        __restrict__ out)  // [M][N]
{
  const int K = I_;
  const int N = H_;
  __shared__ __bf16 lds_b0[128 * 32];   // buffer 0: [n_local*32 + k]
  __shared__ __bf16 lds_b1[128 * 32];   // buffer 1

  const int tid  = threadIdx.x;
  const int lane = tid & 31;
  const int wave = tid >> 5;
  const int wm = wave & 3;                // M sub-tile 0..3 (32 rows each)
  const int wn = wave >> 2;               // N sub-tile 0..1 (64 cols each)
  const int row0 = blockIdx.x * 128 + wm * 32;
  const int col0 = blockIdx.y * 128;

  // --- staging assignment: thread -> (column, 2 x 16B chunks) ---
  const int s_col = tid >> 1;             // 0..127
  const int s_q   = (tid & 1) * 16;       // element offset 0 or 16 within chunk
  const __bf16* s_src = wt + (size_t)(col0 + s_col) * K + s_q;
  __bf16* s_d0 = &lds_b0[s_col * 32 + s_q];
  __bf16* s_d1 = &lds_b1[s_col * 32 + s_q];

  // prefetch first k-chunk into buffer 0
  async_copy16(s_src + 0, s_d0);
  async_copy16(s_src + 8, s_d0 + 8);

  const float* xrow0 = x + (size_t)(row0 + (lane & 15)) * K;
  const float* xrow1 = x + (size_t)(row0 + 16 + (lane & 15)) * K;
  const int kb  = (lane >> 4) * 8;        // A-fragment K half select
  const int kbB = (lane >> 4) * 8;        // B-fragment K half select
  const int nl  = lane & 15;

  const __bf16* b_rd0 = &lds_b0[(wn * 64 + nl) * 32 + kbB];
  const __bf16* b_rd1 = &lds_b1[(wn * 64 + nl) * 32 + kbB];

  v8f acc[2][4];
#pragma unroll
  for (int f = 0; f < 2; ++f)
#pragma unroll
    for (int nt = 0; nt < 4; ++nt)
      acc[f][nt] = (v8f){0.f,0.f,0.f,0.f,0.f,0.f,0.f,0.f};

  for (int k0 = 0; k0 < K; k0 += 64) {
    // ---------- chunk A: buffer 0 holds K [k0, k0+32) ----------
    async_wait0();
    __syncthreads();
    {  // prefetch [k0+32, k0+64) into buffer 1 (always in range: K % 64 == 0)
      const __bf16* ns = s_src + k0 + 32;
      async_copy16(ns + 0, s_d1);
      async_copy16(ns + 8, s_d1 + 8);
    }
    {
      const v16bf a0 = load_a_frag(xrow0, k0, kb);
      const v16bf a1 = load_a_frag(xrow1, k0, kb);
#pragma unroll
      for (int nt = 0; nt < 4; ++nt) {
        v16bf b;
        ((v8bf*)&b)[0] = *(const v8bf*)(b_rd0 + nt * (16 * 32));
        ((v8bf*)&b)[1] = *(const v8bf*)(b_rd0 + nt * (16 * 32) + 16);
        acc[0][nt] = __builtin_amdgcn_wmma_f32_16x16x32_bf16(
            false, a0, false, b, (short)0, acc[0][nt], false, false);
        acc[1][nt] = __builtin_amdgcn_wmma_f32_16x16x32_bf16(
            false, a1, false, b, (short)0, acc[1][nt], false, false);
      }
    }
    // ---------- chunk B: buffer 1 holds K [k0+32, k0+64) ----------
    async_wait0();
    __syncthreads();
    if (k0 + 64 < K) {  // prefetch [k0+64, k0+96) into buffer 0
      const __bf16* ns = s_src + k0 + 64;
      async_copy16(ns + 0, s_d0);
      async_copy16(ns + 8, s_d0 + 8);
    }
    {
      const v16bf a0 = load_a_frag(xrow0, k0 + 32, kb);
      const v16bf a1 = load_a_frag(xrow1, k0 + 32, kb);
#pragma unroll
      for (int nt = 0; nt < 4; ++nt) {
        v16bf b;
        ((v8bf*)&b)[0] = *(const v8bf*)(b_rd1 + nt * (16 * 32));
        ((v8bf*)&b)[1] = *(const v8bf*)(b_rd1 + nt * (16 * 32) + 16);
        acc[0][nt] = __builtin_amdgcn_wmma_f32_16x16x32_bf16(
            false, a0, false, b, (short)0, acc[0][nt], false, false);
        acc[1][nt] = __builtin_amdgcn_wmma_f32_16x16x32_bf16(
            false, a1, false, b, (short)0, acc[1][nt], false, false);
      }
    }
  }

  // ---- epilogue: C layout -> lane&15 = col, VGPR i = row i (+8 hi lanes)
  const int mro = (lane >> 4) * 8;
#pragma unroll
  for (int nt = 0; nt < 4; ++nt) {
    const int n = col0 + wn * 64 + nt * 16 + nl;
    const float bias = bih[n] + bhh[n];
#pragma unroll
    for (int f = 0; f < 2; ++f) {
#pragma unroll
      for (int i = 0; i < 8; ++i) {
        const int m = row0 + f * 16 + i + mro;
        out[(size_t)m * N + n] = acc[f][nt][i] + bias;
      }
    }
  }
}

// =====================================================================
// Kernel B: persistent recurrent scan (h = tanh(xp_t + h @ W_hh)).
//   64 WGs x 128 threads. Each WG owns 16 H-columns; its W_hh slice
//   (16x1024 bf16 = 32KB) is TDM-loaded into LDS once and stays there
//   for all 512 steps. Double-buffered bf16 h state; grid barrier/step.
// =====================================================================
__global__ __launch_bounds__(128)
void k_rnn_steps(const __bf16* __restrict__ whh_t,  // [N=H][K=H] bf16
                 __bf16* __restrict__ h0,
                 __bf16* __restrict__ h1,
                 float*  __restrict__ out,          // [B][T][H] (xp in, h out)
                 float*  __restrict__ hlast,        // [B][H]
                 unsigned* __restrict__ bar)
{
  const int K = H_;
  __shared__ __bf16 lds_w[16 * H_];   // 32 KB

  const int wg  = blockIdx.x;         // owns columns [wg*16, wg*16+16)
  const int tid = threadIdx.x;

#if HAS_TDM
  if (tid < 32) {  // one wave issues the TDM descriptor (EXEC-independent DMA)
    const unsigned lds_base =
        (unsigned)(unsigned long long)(__attribute__((address_space(3))) const void*)lds_w;
    const unsigned long long ga =
        (unsigned long long)(const void*)(whh_t + (size_t)wg * 16 * K);
    u32x4 g0;
    g0[0] = 1u;                                                // count=1 valid D#
    g0[1] = lds_base;                                          // lds_addr
    g0[2] = (unsigned)ga;                                      // global_addr lo
    g0[3] = (unsigned)((ga >> 32) & 0x01FFFFFFu) | (2u << 30); // ga hi | type=2
    i32x8 g1;
    g1[0] = 0x10000;            // data_size=1 (2 bytes/elem)
    g1[1] = (int)(1024u << 16); // tensor_dim0 = 1024 (K)
    g1[2] = (int)(16u << 16);   // tensor_dim1 = 16 rows
    g1[3] = (int)(1024u << 16); // tile_dim0 = 1024
    g1[4] = 16;                 // tile_dim1 = 16
    g1[5] = 1024;               // tensor_dim0_stride = 1024
    g1[6] = 0;
    g1[7] = 0;
    i32x4 z4 = {0, 0, 0, 0};
    i32x8 z8 = {0, 0, 0, 0, 0, 0, 0, 0};
    __builtin_amdgcn_tensor_load_to_lds(g0, g1, z4, z4, z8, 0);
    __builtin_amdgcn_s_wait_tensorcnt(0);
  }
#else
  {
    const v8bf* src = (const v8bf*)(whh_t + (size_t)wg * 16 * K);
    v8bf* dst = (v8bf*)lds_w;
    for (int i = tid; i < (16 * K) / 8; i += 128) dst[i] = src[i];
  }
#endif
  __syncthreads();

  const int lane = tid & 31;
  const int wave = tid >> 5;           // 0..3 -> batch-row tile
  const int m0   = wave * 16;
  const int mA   = m0 + (lane & 15);
  const int kb   = (lane >> 4) * 8;
  const int nl   = lane & 15;
  const int kbB  = (lane >> 4) * 8;
  const int n    = wg * 16 + nl;
  const int mro  = (lane >> 4) * 8;
  const unsigned nwg = gridDim.x;

  for (int t = 0; t < T_; ++t) {
    const __bf16* hr = (t & 1) ? h1 : h0;
    __bf16*       hw = (t & 1) ? h0 : h1;

    v8f acc = (v8f){0.f,0.f,0.f,0.f,0.f,0.f,0.f,0.f};
    const __bf16* hrow = hr + (size_t)mA * K;
    for (int k0 = 0; k0 < K; k0 += 32) {
      v16bf a;
      ((v8bf*)&a)[0] = *(const v8bf*)(hrow + k0 + kb);
      ((v8bf*)&a)[1] = *(const v8bf*)(hrow + k0 + 16 + kb);
      v16bf b;
      ((v8bf*)&b)[0] = *(const v8bf*)(lds_w + nl * K + k0 + kbB);
      ((v8bf*)&b)[1] = *(const v8bf*)(lds_w + nl * K + k0 + 16 + kbB);
      acc = __builtin_amdgcn_wmma_f32_16x16x32_bf16(
          false, a, false, b, (short)0, acc, false, false);
    }

#pragma unroll
    for (int i = 0; i < 8; ++i) {
      const int m = m0 + i + mro;
      const size_t oidx = ((size_t)m * T_ + t) * H_ + n;
      const float v = tanhf(acc[i] + out[oidx]);
      out[oidx] = v;
      hw[(size_t)m * H_ + n] = (__bf16)v;
      if (t == T_ - 1) hlast[(size_t)m * H_ + n] = v;
    }

    // ---- grid-wide barrier (monotonic phase counter) ----
    __threadfence();
    __syncthreads();
    if (tid == 0) {
      __hip_atomic_fetch_add(bar, 1u, __ATOMIC_RELEASE, __HIP_MEMORY_SCOPE_AGENT);
      const unsigned target = (unsigned)(t + 1) * nwg;
      while (__hip_atomic_load(bar, __ATOMIC_ACQUIRE, __HIP_MEMORY_SCOPE_AGENT) < target)
        __builtin_amdgcn_s_sleep(1);
    }
    __syncthreads();
  }
}

// =====================================================================
// Launch
// =====================================================================
extern "C" void kernel_launch(void* const* d_in, const int* in_sizes, int n_in,
                              void* d_out, int out_size, void* d_ws, size_t ws_size,
                              hipStream_t stream) {
  (void)in_sizes; (void)n_in; (void)out_size; (void)ws_size;

  const float* x    = (const float*)d_in[0];  // [B,T,I]
  const float* hx   = (const float*)d_in[1];  // [B,H]
  const float* wih  = (const float*)d_in[2];  // [I,H]
  const float* bih  = (const float*)d_in[3];  // [H]
  const float* whh  = (const float*)d_in[4];  // [H,H]
  const float* bhh  = (const float*)d_in[5];  // [H]

  float* out   = (float*)d_out;
  float* hlast = out + (size_t)B_ * T_ * H_;

  char* ws = (char*)d_ws;
  __bf16*   wih_t = (__bf16*)(ws);                                    // 2 MB
  __bf16*   whh_t = (__bf16*)(ws + (2u << 20));                       // 2 MB
  __bf16*   h0    = (__bf16*)(ws + (4u << 20));                       // 128 KB
  __bf16*   h1    = (__bf16*)(ws + (4u << 20) + (128u << 10));        // 128 KB
  unsigned* bar   = (unsigned*)(ws + (4u << 20) + (256u << 10));      // 64 B

  (void)hipMemsetAsync(bar, 0, 64, stream);

  k_transpose_bf16<<<(I_ * H_ + 255) / 256, 256, 0, stream>>>(wih, wih_t, I_, H_);
  k_transpose_bf16<<<(H_ * H_ + 255) / 256, 256, 0, stream>>>(whh, whh_t, H_, H_);
  k_hx_to_bf16<<<(B_ * H_ + 255) / 256, 256, 0, stream>>>(hx, h0);

  dim3 gA((B_ * T_) / 128, H_ / 128);   // 256 x 8 blocks of 256 threads
  k_gemm_xproj<<<gA, 256, 0, stream>>>(x, wih_t, bih, bhh, out);

  k_rnn_steps<<<64, 128, 0, stream>>>(whh_t, h0, h1, out, hlast, bar);
}
